// SpikingNeuralNetwork_81338090652085
// MI455X (gfx1250) — matrix operand order repaired and checked by reading, not compiled, not run on hardware
//
#include <hip/hip_runtime.h>

// ---------------------------------------------------------------------------
// SNN: latency-encode -> 3x (GEMM + LIF scan), fused per layer.
// MI455X reasoning: ~4.3 GFLOP of GEMM vs ~40 MB traffic @ 23.3 TB/s ->
// bandwidth/latency bound, and with only ~512 waves the device cannot hide
// memory latency by oversubscription. So: f16 activations, WMMA
// f32_16x16x32_f16 for the matmuls, membrane state held in the D-accumulator
// layout across the 32-step scan, weight B-tiles preloaded to VGPRs, and the
// per-timestep A tiles software-pipelined (ping-pong double buffer) so L2
// latency overlaps the WMMA chain + LIF update of the previous step.
// ---------------------------------------------------------------------------

typedef __attribute__((ext_vector_type(16))) _Float16 v16h;
typedef __attribute__((ext_vector_type(8)))  _Float16 v8h;
typedef __attribute__((ext_vector_type(8)))  float    v8f;

#define T_STEPS 32
#define BATCH   512
#define F_IN    128
#define SEQ     64

// ---------------------------------------------------------------------------
// Kernel 1: latency encoding + mean over sequence dim.
// One thread per (b,f). Two passes over s (min/max, then time index); the
// 64 time indices are packed as bytes into 16 u32s (register-resident,
// all indices compile-time constant via full unroll), then the 32 output
// counts are produced by compare-accumulate.
// ---------------------------------------------------------------------------
__global__ __launch_bounds__(256)
void latency_encode_kernel(const float* __restrict__ x, _Float16* __restrict__ out)
{
    const int gid = blockIdx.x * blockDim.x + threadIdx.x;   // 0 .. 65535
    const int f = gid & (F_IN - 1);
    const int b = gid >> 7;
    const float* px = x + (size_t)b * (SEQ * F_IN) + f;

    // pass 1: gated min/max over s
    float mn = 3.4e38f, mx = -3.4e38f;
    #pragma unroll
    for (int s = 0; s < SEQ; ++s) {
        float v = px[s * F_IN];
        v = (v < 0.75f) ? 0.0f : v;          // ENC_THR gate
        mn = fminf(mn, v);
        mx = fmaxf(mx, v);
    }
    const float inv = 1.0f / (mx - mn + 1e-8f);

    // t_max = log((0.01+1e-7)/1e-7) = 11.512935464920228 (float64, as in numpy)
    const float SCALE = 31.0f / 11.512935464920228f;

    unsigned pk[16];
    #pragma unroll
    for (int j = 0; j < 16; ++j) pk[j] = 0u;

    // pass 2: per-element spike time index, packed into bytes
    #pragma unroll
    for (int s = 0; s < SEQ; ++s) {
        float v = px[s * F_IN];
        v = (v < 0.75f) ? 0.0f : v;
        float xn = (v - mn) * inv;                       // minmax normalize
        float d  = fmaxf(xn, 0.01f + 1e-7f);             // clip at LAT_THR+EPS
        float tt = __logf(d / (d - 0.01f)) * SCALE;      // log latency code
        int ti = (int)rintf(tt);                         // RNE == jnp.round
        ti = ti < 0 ? 0 : (ti > 31 ? 31 : ti);
        pk[s >> 2] |= (unsigned)ti << ((s & 3) * 8);
    }

    // histogram over time -> mean over s (count / 64), stored f16
    for (int t = 0; t < T_STEPS; ++t) {
        int c = 0;
        #pragma unroll
        for (int j = 0; j < 16; ++j) {
            unsigned w = pk[j];
            c += (int)((w & 0xFFu)         == (unsigned)t);
            c += (int)(((w >> 8) & 0xFFu)  == (unsigned)t);
            c += (int)(((w >> 16) & 0xFFu) == (unsigned)t);
            c += (int)((w >> 24)           == (unsigned)t);
        }
        out[(((size_t)t * BATCH) + b) * F_IN + f] = (_Float16)((float)c * 0.015625f);
    }
}

// ---------------------------------------------------------------------------
// Weight f32 -> f16 conversion (weights are tiny; done once per launch).
// ---------------------------------------------------------------------------
__global__ __launch_bounds__(256)
void f32_to_f16_kernel(const float* __restrict__ s, _Float16* __restrict__ d, int n)
{
    int i = blockIdx.x * blockDim.x + threadIdx.x;
    if (i < n) d[i] = (_Float16)s[i];
}

// ---------------------------------------------------------------------------
// Kernel 2: fused GEMM + LIF scan for one SNN block.
// Each wave owns one 16(b) x 16(h) tile and loops t = 0..31:
//   cur = A(t) x W^T + bias  via K/32 v_wmma_f32_16x16x32_f16 steps,
//   mem = beta*mem + cur - reset*thr  (mem lives in the D-accumulator layout),
//   spk = (mem - thr > 0)  stored as f16 (or f32 for the final block).
// A tiles are double-buffered across t (prefetch t+1 while computing t);
// weight B-tiles are preloaded to VGPRs once (reused 32x over t).
// Grids are sized exactly -> EXEC is all-ones everywhere (WMMA requirement).
// ---------------------------------------------------------------------------
template<int K, int N, bool F32OUT>
__global__ __launch_bounds__(128)
void snn_gemm_lif_kernel(const _Float16* __restrict__ Ain,  // (T*BATCH) x K, row = t*BATCH+b
                         const _Float16* __restrict__ Wh,   // N x K row-major (f16)
                         const float*    __restrict__ bias, // N (f32)
                         _Float16* __restrict__ SpkH,       // (T*BATCH) x N, f16 (if !F32OUT)
                         float*    __restrict__ SpkF)       // (T*BATCH) x N, f32 (if F32OUT)
{
    constexpr int KT = K / 32;
    const int wave   = blockIdx.x * (blockDim.x >> 5) + (threadIdx.x >> 5);
    const int hTiles = N / 16;
    const int bTile  = wave / hTiles;
    const int hTile  = wave % hTiles;
    const int lane   = threadIdx.x & 31;
    const int half   = lane >> 4;      // 0: lanes 0-15, 1: lanes 16-31
    const int l16    = lane & 15;

    // ---- preload B (= W^T) tiles into registers ----
    // B element (k, n): lane holds col n = lane%16; k-halves {0..7,16..23} (half 0)
    // or {8..15,24..31} (half 1) -> two contiguous 8-half reads of a W row.
    v16h breg[KT];
    const _Float16* wp = Wh + (size_t)(hTile * 16 + l16) * K + half * 8;
    #pragma unroll
    for (int kk = 0; kk < KT; ++kk) {
        v8h lo = *(const v8h*)(wp + kk * 32);
        v8h hi = *(const v8h*)(wp + kk * 32 + 16);
        #pragma unroll
        for (int i = 0; i < 8; ++i) { breg[kk][i] = lo[i]; breg[kk][8 + i] = hi[i]; }
    }

    const float biasn = bias[hTile * 16 + l16];   // col n == l16, same for all 8 rows

    // A: lane holds row M = lane%16; k-halves {0..7,16..23} (half 0) or
    // {8..15,24..31} (half 1) -> two contiguous 16-byte loads per K-step.
    const _Float16* arowbase = Ain + ((size_t)bTile * 16 + l16) * K + half * 8;

    // Load the full A tile for one timestep into a register array (issues all
    // 2*KT b128 loads as one clause instead of one wait per WMMA).
    auto loadA = [&](v16h* dst, int t) {
        const _Float16* ap = arowbase + (size_t)t * BATCH * K;
        #pragma unroll
        for (int kk = 0; kk < KT; ++kk) {
            v8h lo = *(const v8h*)(ap + kk * 32);
            v8h hi = *(const v8h*)(ap + kk * 32 + 16);
            #pragma unroll
            for (int i = 0; i < 8; ++i) { dst[kk][i] = lo[i]; dst[kk][8 + i] = hi[i]; }
        }
    };

    v8f mem = {};   // membrane potential, D-accumulator layout, zeros at t=0

    // WMMA chain + LIF update + spike store for one timestep.
    auto stepLIF = [&](const v16h* a, int t) {
        v8f acc = {};
        #pragma unroll
        for (int kk = 0; kk < KT; ++kk) {
            acc = __builtin_amdgcn_wmma_f32_16x16x32_f16(
                      /*neg_a=*/false, a[kk], /*neg_b=*/false, breg[kk],
                      /*c_mod=*/(short)0, acc, /*reuse_a=*/false, /*reuse_b=*/false);
        }
        // D layout: VGPR r -> row m = r + 8*half, col n = l16.
        const size_t outb = (((size_t)t * BATCH) + bTile * 16 + half * 8) * N
                            + hTile * 16 + l16;
        #pragma unroll
        for (int r = 0; r < 8; ++r) {
            float cur   = acc[r] + biasn;
            float reset = (mem[r] > 1.0f) ? 1.0f : 0.0f;   // reset*thr from prev mem
            float m     = 0.9f * mem[r] + cur - reset;      // beta=0.9
            mem[r]      = m;
            float spk   = (m > 1.0f) ? 1.0f : 0.0f;         // heaviside(mem - thr)
            if (F32OUT) SpkF[outb + (size_t)r * N] = spk;
            else        SpkH[outb + (size_t)r * N] = (_Float16)spk;
        }
    };

    // Software pipeline: ping-pong A buffers across timesteps so the next
    // step's loads overlap the current step's WMMAs + LIF VALU work.
    v16h a0[KT], a1[KT];
    loadA(a0, 0);
    for (int t = 0; t < T_STEPS; t += 2) {
        loadA(a1, t + 1);
        stepLIF(a0, t);
        loadA(a0, (t + 2) & (T_STEPS - 1));   // wraps to 0 on last iter (in-bounds, branch-free)
        stepLIF(a1, t + 1);
    }
}

// ---------------------------------------------------------------------------
// Launch. Inputs: x, W1, b1, W2, b2, W3, b3 (all f32). Output: (32,512,128) f32.
// Workspace: enc f16 4MB | spk1 f16 8MB | spk2 f16 8MB | f16 weights 256KB.
// ---------------------------------------------------------------------------
extern "C" void kernel_launch(void* const* d_in, const int* in_sizes, int n_in,
                              void* d_out, int out_size, void* d_ws, size_t ws_size,
                              hipStream_t stream)
{
    const float* x  = (const float*)d_in[0];
    const float* W1 = (const float*)d_in[1];
    const float* b1 = (const float*)d_in[2];
    const float* W2 = (const float*)d_in[3];
    const float* b2 = (const float*)d_in[4];
    const float* W3 = (const float*)d_in[5];
    const float* b3 = (const float*)d_in[6];
    float* out = (float*)d_out;

    char* ws = (char*)d_ws;
    _Float16* enc  = (_Float16*)(ws);                      // 32*512*128 = 2M halves
    _Float16* spk1 = (_Float16*)(ws + (4u  << 20));        // 32*512*256 = 4M halves
    _Float16* spk2 = (_Float16*)(ws + (12u << 20));        // 32*512*256
    _Float16* w1h  = (_Float16*)(ws + (20u << 20));        // 256*128
    _Float16* w2h  = w1h + 256 * 128;                      // 256*256
    _Float16* w3h  = w2h + 256 * 256;                      // 128*256

    f32_to_f16_kernel<<<(256 * 128 + 255) / 256, 256, 0, stream>>>(W1, w1h, 256 * 128);
    f32_to_f16_kernel<<<(256 * 256 + 255) / 256, 256, 0, stream>>>(W2, w2h, 256 * 256);
    f32_to_f16_kernel<<<(128 * 256 + 255) / 256, 256, 0, stream>>>(W3, w3h, 128 * 256);

    // 512*128 threads, one per (b,f)
    latency_encode_kernel<<<256, 256, 0, stream>>>(x, enc);

    // Block 1: K=128, N=256 -> 32*16 = 512 waves = 128 blocks of 4 waves
    snn_gemm_lif_kernel<128, 256, false><<<128, 128, 0, stream>>>(enc,  w1h, b1, spk1, nullptr);
    // Block 2: K=256, N=256 -> 512 waves
    snn_gemm_lif_kernel<256, 256, false><<<128, 128, 0, stream>>>(spk1, w2h, b2, spk2, nullptr);
    // Block 3: K=256, N=128 -> 256 waves = 64 blocks; writes f32 output
    snn_gemm_lif_kernel<256, 128, true ><<< 64, 128, 0, stream>>>(spk2, w3h, b3, nullptr, out);
}
